// NEURONSv2Layer_61950608277919
// MI455X (gfx1250) — matrix operand order repaired and checked by reading, not compiled, not run on hardware
//
#include <hip/hip_runtime.h>
#include <stdint.h>

typedef __attribute__((ext_vector_type(16))) __bf16 v16bf;
typedef __attribute__((ext_vector_type(8)))  float  v8f;

#define B_ 32
#define T_ 4096
#define I_ 256
#define O_ 256
#define R_ (B_ * T_)          // 131072 rows total
#define XS_PAD 8
#define WS_PAD 8
#define NCHUNK 56             // 7 matrices * 8 K-chunks

__device__ __forceinline__ unsigned short f32_to_bf16(float f) {
    union { float f; unsigned u; } v; v.f = f;
    unsigned u = v.u;
    unsigned r = u + 0x7FFFu + ((u >> 16) & 1u);   // round-to-nearest-even
    return (unsigned short)(r >> 16);
}

__device__ __forceinline__ uint32_t pack2_bf16(float a, float b) {
    return (uint32_t)f32_to_bf16(a) | ((uint32_t)f32_to_bf16(b) << 16);
}

__device__ __forceinline__ v8f v8f_zero() {
    v8f z;
#pragma unroll
    for (int i = 0; i < 8; ++i) z[i] = 0.0f;
    return z;
}

__device__ __forceinline__ const float* wm_base(const float* Wsoma,
                                                const float* Wbasal,
                                                const float* Wapical, int m) {
    return (m < 4) ? (Wbasal + (size_t)m * (O_ * I_))
         : (m < 6) ? (Wapical + (size_t)(m - 4) * (O_ * I_))
                   : Wsoma;
}

// Stage one 256x32 f32 weight slab into an LDS bf16 buffer (thread tid -> col tid).
__device__ __forceinline__ void stage_w(const float* __restrict__ src,
                                        unsigned short (*__restrict__ wsbuf)[32 + WS_PAD],
                                        int tid) {
#pragma unroll
    for (int i = 0; i < 32; i += 8) {
        float4 f0 = *(const float4*)(src + i);
        float4 f1 = *(const float4*)(src + i + 4);
        uint4 p;
        p.x = pack2_bf16(f0.x, f0.y);
        p.y = pack2_bf16(f0.z, f0.w);
        p.z = pack2_bf16(f1.x, f1.y);
        p.w = pack2_bf16(f1.z, f1.w);
        *(uint4*)&wsbuf[tid][i] = p;
    }
}

// ---------------------------------------------------------------------------
// Phase 1: batched GEMM over all (b,t) rows.
//   Sout[r,o] = x_r . W_soma[o,:]
//   Dout[r,o] = 0.25 * sum_{n<4} relu(x_r . W_basal[n,o,:])
//             + 0.50 * sum_{n<2} relu(x_r . W_apical[n,o,:])
// bf16 inputs, f32 accumulation via v_wmma_f32_16x16x32_bf16.
// Workgroup tile: 32 rows x 256 cols; 8 waves, each wave owns a 16-row half
// and a 64-col strip (4 WMMA tiles). W slabs double-buffered in LDS; the
// flattened 56-chunk pipeline overlaps global staging with WMMA.
// ---------------------------------------------------------------------------
__global__ __launch_bounds__(256)
void neurons_gemm_kernel(const float* __restrict__ X,
                         const float* __restrict__ Wsoma,
                         const float* __restrict__ Wbasal,
                         const float* __restrict__ Wapical,
                         float* __restrict__ Sout,
                         float* __restrict__ Dout)
{
    __shared__ __attribute__((aligned(16))) unsigned short Xs[32][I_ + XS_PAD];
    __shared__ __attribute__((aligned(16))) unsigned short Ws[2][O_][32 + WS_PAD];

    const int tid     = threadIdx.x;
    const int lane    = tid & 31;
    const int wave    = tid >> 5;
    const int rowbase = blockIdx.x * 32;

    // ---- stage X tile [32 x 256] as bf16 into LDS (once) ----
    {
        const int row = tid >> 3;            // 32 rows, 8 threads per row
        const int cb  = (tid & 7) * 32;      // 32 contiguous cols per thread
        const float* src = X + (size_t)(rowbase + row) * I_ + cb;
#pragma unroll
        for (int i = 0; i < 32; i += 8) {
            float4 f0 = *(const float4*)(src + i);
            float4 f1 = *(const float4*)(src + i + 4);
            uint4 p;
            p.x = pack2_bf16(f0.x, f0.y);
            p.y = pack2_bf16(f0.z, f0.w);
            p.z = pack2_bf16(f1.x, f1.y);
            p.w = pack2_bf16(f1.z, f1.w);
            *(uint4*)&Xs[row][cb + i] = p;
        }
    }

    // Prime the W pipeline: chunk 0 -> buffer 0.
    stage_w(wm_base(Wsoma, Wbasal, Wapical, 0) + (size_t)tid * I_,
            Ws[0], tid);
    __syncthreads();

    const int rt    = wave >> 2;            // 0..1 : which 16-row half
    const int cb64  = (wave & 3) * 64;      // 0/64/128/192 : 64-col strip
    const int arow  = rt * 16 + (lane & 15);
    const int ahalf = lane >> 4;            // half-wave selector
    const int ncol0 = cb64 + (lane & 15);

    v8f Dacc[4], acc[4];
#pragma unroll
    for (int ct = 0; ct < 4; ++ct) { Dacc[ct] = v8f_zero(); acc[ct] = v8f_zero(); }

    for (int q = 0; q < NCHUNK; ++q) {
        const int kc = (q & 7) * 32;

        // Stage next slab into the other buffer while we compute this one.
        if (q < NCHUNK - 1) {
            const int qn = q + 1;
            const float* src = wm_base(Wsoma, Wbasal, Wapical, qn >> 3)
                             + (size_t)tid * I_ + (qn & 7) * 32;
            stage_w(src, Ws[qn & 1], tid);
        }

        // A fragment: 16x32 bf16. Lane holds K = kc + half*8 + {0..7, 16..23}
        union { v16bf v; uint4 q4[2]; } ua;
        {
            const unsigned short* xr = &Xs[arow][kc + ahalf * 8];
            ua.q4[0] = *(const uint4*)(xr);        // K offsets 0..7
            ua.q4[1] = *(const uint4*)(xr + 16);   // K offsets 16..23
        }

        // All four B fragments first, then back-to-back WMMAs.
        union { v16bf v; uint4 q4[2]; } ub[4];
#pragma unroll
        for (int ct = 0; ct < 4; ++ct) {
            const unsigned short* wr = &Ws[q & 1][ncol0 + ct * 16][ahalf * 8];
            ub[ct].q4[0] = *(const uint4*)(wr);      // K offsets 0..7
            ub[ct].q4[1] = *(const uint4*)(wr + 16); // K offsets 16..23
        }
#pragma unroll
        for (int ct = 0; ct < 4; ++ct) {
            acc[ct] = __builtin_amdgcn_wmma_f32_16x16x32_bf16(
                false, ua.v, false, ub[ct].v, (short)0, acc[ct], false, false);
        }

        if ((q & 7) == 7) {
            const int m = q >> 3;
            if (m < 6) {
                const float coef = (m < 4) ? 0.25f : 0.5f;
#pragma unroll
                for (int ct = 0; ct < 4; ++ct) {
#pragma unroll
                    for (int r = 0; r < 8; ++r) {
                        float v = acc[ct][r];
                        Dacc[ct][r] += coef * (v > 0.0f ? v : 0.0f);
                    }
                    acc[ct] = v8f_zero();
                }
            } else {
                // soma pass: emit S and accumulated D
#pragma unroll
                for (int ct = 0; ct < 4; ++ct) {
#pragma unroll
                    for (int r = 0; r < 8; ++r) {
                        const int row = rowbase + rt * 16 + r + (lane >> 4) * 8;
                        const int col = cb64 + ct * 16 + (lane & 15);
                        const size_t idx = (size_t)row * O_ + col;
                        Sout[idx] = acc[ct][r];
                        Dout[idx] = Dacc[ct][r];
                    }
                }
            }
        }
        __syncthreads();   // one barrier per chunk: staging <-> consumption fence
    }
}

// ---------------------------------------------------------------------------
// Phase 2: sequential scan over T. One block per batch (b), one thread per
// neuron (o). Reads S from the spikes slot and D from the phases slot of
// d_out, overwrites them in place with spikes/phases; writes membrane.
// ---------------------------------------------------------------------------
__global__ __launch_bounds__(256)
void neurons_scan_kernel(float* __restrict__ spikes,    // in: S,  out: spikes
                         float* __restrict__ phases,    // in: D,  out: theta
                         float* __restrict__ membrane)  // out: mem
{
    const int b    = blockIdx.x;
    const int o    = threadIdx.x;
    const int lane = o & 31;
    const int wid  = o >> 5;

    __shared__ float red_c[8];
    __shared__ float red_s[8];

    const float a_mem  = 0.951229424500714f;   // exp(-1/20)
    const float a_syn  = 0.818730753077982f;   // exp(-1/5)
    const float a_dend = 0.904837418035960f;   // exp(-1/10)
    const float TWO_PI = 6.28318530717958647692f;

    float mem = 0.0f, syn = 0.0f, dend = 0.0f, theta = 0.0f;
    const size_t base = (size_t)b * T_ * O_ + o;

    for (int t = 0; t < T_; ++t) {
        const size_t idx = base + (size_t)t * O_;
        const float S    = spikes[idx];
        const float Draw = phases[idx];
        if (t + 1 < T_) {
            __builtin_prefetch(&spikes[idx + O_], 0, 0);
            __builtin_prefetch(&phases[idx + O_], 0, 0);
        }

        // mean-field over O=256 oscillators (pre-update theta)
        float c = __cosf(theta);
        float s = __sinf(theta);
#pragma unroll
        for (int off = 16; off > 0; off >>= 1) {
            c += __shfl_xor(c, off, 32);
            s += __shfl_xor(s, off, 32);
        }
        if (lane == 0) { red_c[wid] = c; red_s[wid] = s; }
        __syncthreads();
        float csum = 0.0f, ssum = 0.0f;
#pragma unroll
        for (int w = 0; w < 8; ++w) { csum += red_c[w]; ssum += red_s[w]; }
        __syncthreads();
        const float cm  = csum * (1.0f / 256.0f);
        const float sm  = ssum * (1.0f / 256.0f);
        const float r   = sqrtf(cm * cm + sm * sm);
        const float psi = atan2f(sm, cm);

        // LIF dynamics
        dend = a_dend * dend + (1.0f - a_dend) * Draw;
        const float drive = S + 0.5f * dend;
        syn = a_syn * syn + drive;
        mem = a_mem * mem + syn;
        const float v     = mem - 1.0f;
        const float spike = (v > 0.0f) ? 1.0f : 0.0f;
        mem -= spike;

        // Kuramoto phase update
        const float dtheta = TWO_PI * 10.0f + r * __sinf(psi - theta) + spike;
        theta = fmodf(theta + 0.001f * dtheta, TWO_PI);

        spikes[idx] = spike;
        phases[idx] = theta;
        __builtin_nontemporal_store(mem, &membrane[idx]);  // write-once stream
    }
}

extern "C" void kernel_launch(void* const* d_in, const int* in_sizes, int n_in,
                              void* d_out, int out_size, void* d_ws, size_t ws_size,
                              hipStream_t stream) {
    (void)in_sizes; (void)n_in; (void)out_size; (void)d_ws; (void)ws_size;

    const float* X       = (const float*)d_in[0];  // [B,T,I]
    const float* Wsoma   = (const float*)d_in[1];  // [O,I]
    const float* Wbasal  = (const float*)d_in[2];  // [4,O,I]
    const float* Wapical = (const float*)d_in[3];  // [2,O,I]

    float* out      = (float*)d_out;
    float* spikes   = out;                               // staging: S
    float* phases   = out + (size_t)B_ * T_ * O_;        // staging: D
    float* membrane = out + 2 * (size_t)B_ * T_ * O_;

    neurons_gemm_kernel<<<R_ / 32, 256, 0, stream>>>(
        X, Wsoma, Wbasal, Wapical, spikes, phases);
    neurons_scan_kernel<<<B_, 256, 0, stream>>>(spikes, phases, membrane);
}
